// NonLocalFeatures_16887811408667
// MI455X (gfx1250) — compile-verified, matrix operand order checked by reading, not run on hardware
//
#include <hip/hip_runtime.h>
#include <hip/hip_bf16.h>

typedef __attribute__((ext_vector_type(16))) _Float16 v16h;
typedef __attribute__((ext_vector_type(8)))  float    v8f;

#define B_   4
#define N_   8192
#define K_   32
#define H_   128
#define NPTS (B_ * N_)

#define WAVES_PER_WG  8
#define PTS_PER_WAVE  8
#define LDS_STRIDE_H  136   // halfs per row (272 B) -> conflict-free b128 LDS reads

struct alignas(8) Half4 { _Float16 a, b, c, d; };

// ---------------------------------------------------------------------------
// Kernel 0: Wt[h][c] = (f16) W[c][h]  for c in [0,128)   (WMMA B operand layout)
// ---------------------------------------------------------------------------
__global__ void NLF_wt_kernel(const float* __restrict__ W, _Float16* __restrict__ Wt) {
    int g = blockIdx.x * blockDim.x + threadIdx.x;
    if (g >= H_ * H_) return;
    int h = g >> 7, c = g & 127;
    Wt[h * H_ + c] = (_Float16)W[c * H_ + h];
}

// ---------------------------------------------------------------------------
// Kernel 1: LayerNorm each [128] row, emit f16. One wave32 per row.
// ---------------------------------------------------------------------------
__global__ __launch_bounds__(256) void NLF_ln_kernel(
    const float* __restrict__ x, const float* __restrict__ gamma,
    const float* __restrict__ beta, _Float16* __restrict__ y) {
    int wave = (blockIdx.x * blockDim.x + threadIdx.x) >> 5;
    int lane = threadIdx.x & 31;
    if (wave >= NPTS) return;

    const float4* row = (const float4*)(x + (size_t)wave * H_);
    float4 v = row[lane];
    float s  = v.x + v.y + v.z + v.w;
    float s2 = v.x * v.x + v.y * v.y + v.z * v.z + v.w * v.w;
    for (int m = 16; m >= 1; m >>= 1) {
        s  += __shfl_xor(s,  m, 32);
        s2 += __shfl_xor(s2, m, 32);
    }
    float mu  = s * (1.0f / H_);
    float var = s2 * (1.0f / H_) - mu * mu;
    float inv = rsqrtf(var + 1e-5f);

    float4 gg = ((const float4*)gamma)[lane];
    float4 bb = ((const float4*)beta)[lane];
    Half4 o;
    o.a = (_Float16)((v.x - mu) * inv * gg.x + bb.x);
    o.b = (_Float16)((v.y - mu) * inv * gg.y + bb.y);
    o.c = (_Float16)((v.z - mu) * inv * gg.z + bb.z);
    o.d = (_Float16)((v.w - mu) * inv * gg.w + bb.w);
    ((Half4*)(y + (size_t)wave * H_))[lane] = o;
}

// ---------------------------------------------------------------------------
// Kernel 2: per point p = (b,n): gather 32 neighbor rows (f16), WMMA against
// Wt (LDS), f32 epilogue: + rel0*W[128] + rel1*W[129] + bias, relu, mean over
// K=32, + residual.
// ---------------------------------------------------------------------------
__global__ __launch_bounds__(256) void NLF_gemm_kernel(
    const _Float16* __restrict__ lnf,   // [B*N][128] f16
    const _Float16* __restrict__ Wt,    // [128][128] f16, h-major
    const float*    __restrict__ W,     // original [130][128] f32
    const float*    __restrict__ bias,  // [128]
    const float*    __restrict__ rel,   // [B*N][32][2]
    const int*      __restrict__ nidx,  // [B*N][32]
    const float*    __restrict__ xin,   // [B*N][128]
    float*          __restrict__ out)   // [B*N][128]
{
    __shared__ _Float16 lw[H_ * LDS_STRIDE_H];

    // Cooperative stage of Wt into LDS (dword granularity, padded stride).
    {
        const uint32_t* src = (const uint32_t*)Wt;
        uint32_t*       dst = (uint32_t*)lw;
        for (int i = threadIdx.x; i < H_ * 64; i += 256) {
            int r = i >> 6, c = i & 63;
            dst[r * (LDS_STRIDE_H / 2) + c] = src[i];
        }
    }
    __syncthreads();

    const int wave   = threadIdx.x >> 5;
    const int lane   = threadIdx.x & 31;
    const int laneM  = lane & 15;   // row (A) / col (B,C) within tile
    const int laneHi = lane >> 4;   // K-group / M-half selector

    const float* w128 = W + 128 * H_;
    const float* w129 = W + 129 * H_;

    union AF { v16h v; float4 f4[2]; };
    union BF { v16h v; float4 f4[2]; };

    for (int p0 = 0; p0 < PTS_PER_WAVE; ++p0) {
        int pt = (blockIdx.x * WAVES_PER_WG + wave) * PTS_PER_WAVE + p0;
        if (pt >= NPTS) break;   // wave-uniform

        const int*      idxp = nidx + (size_t)pt * K_;
        const int       batch = pt / N_;
        const _Float16* lnb   = lnf + (size_t)batch * N_ * H_;

        // ---- A fragments: 2 M-tiles x 4 k-steps of 16x32 f16 -------------
        AF a[2][4];
        float r0v[2][8], r1v[2][8];
        for (int mt = 0; mt < 2; ++mt) {
            int nb = idxp[mt * 16 + laneM];
            const float4* rowp = (const float4*)(lnb + (size_t)nb * H_);
            // lane holds K = laneHi*8 .. +7 and 16+laneHi*8 .. +7 (per ISA layout)
            for (int kk = 0; kk < 4; ++kk) {
                a[mt][kk].f4[0] = rowp[kk * 4 + laneHi];
                a[mt][kk].f4[1] = rowp[kk * 4 + laneHi + 2];
            }
            // relative channels for this lane's epilogue rows: m = mt*16 + 8*laneHi + r
            const float* relp = rel + ((size_t)pt * K_ + mt * 16 + 8 * laneHi) * 2;
            for (int r = 0; r < 8; ++r) {
                r0v[mt][r] = relp[2 * r + 0];
                r1v[mt][r] = relp[2 * r + 1];
            }
        }

        // ---- 8 N-tiles of 16 output channels ----------------------------
        for (int nt = 0; nt < 8; ++nt) {
            int hcol = nt * 16 + laneM;
            const float4* brow = (const float4*)(lw + hcol * LDS_STRIDE_H);
            BF b[4];
            for (int kk = 0; kk < 4; ++kk) {
                b[kk].f4[0] = brow[kk * 4 + laneHi];
                b[kk].f4[1] = brow[kk * 4 + laneHi + 2];
            }

            v8f acc0 = {}; v8f acc1 = {};
            for (int kk = 0; kk < 4; ++kk) {
                acc0 = __builtin_amdgcn_wmma_f32_16x16x32_f16(
                    false, a[0][kk].v, false, b[kk].v, (short)0, acc0, false, false);
                acc1 = __builtin_amdgcn_wmma_f32_16x16x32_f16(
                    false, a[1][kk].v, false, b[kk].v, (short)0, acc1, false, false);
            }

            // ---- epilogue: rank-2 tail + bias + relu + sum over 32 rows --
            float wa = w128[hcol], wb = w129[hcol], bs = bias[hcol];
            float ssum = 0.0f;
            for (int r = 0; r < 8; ++r) {
                float v0 = acc0[r] + r0v[0][r] * wa + r1v[0][r] * wb + bs;
                float v1 = acc1[r] + r0v[1][r] * wa + r1v[1][r] * wb + bs;
                ssum += fmaxf(v0, 0.0f) + fmaxf(v1, 0.0f);
            }
            // C layout: lanes L and L^16 hold the same column, different row halves
            ssum += __shfl_xor(ssum, 16, 32);
            if (lane < 16) {
                size_t o = (size_t)pt * H_ + nt * 16 + lane;
                out[o] = ssum * (1.0f / K_) + xin[o];
            }
        }
    }
}

// ---------------------------------------------------------------------------
extern "C" void kernel_launch(void* const* d_in, const int* in_sizes, int n_in,
                              void* d_out, int out_size, void* d_ws, size_t ws_size,
                              hipStream_t stream) {
    const float* inputs   = (const float*)d_in[0];
    const float* relative = (const float*)d_in[1];
    const float* gamma    = (const float*)d_in[2];
    const float* beta     = (const float*)d_in[3];
    const float* W        = (const float*)d_in[4];
    const float* bias     = (const float*)d_in[5];
    const int*   nidx     = (const int*)d_in[6];
    float*       out      = (float*)d_out;

    _Float16* lnf = (_Float16*)d_ws;                 // [B*N*128] f16  (8 MB)
    _Float16* wt  = lnf + (size_t)NPTS * H_;         // [128*128] f16  (32 KB)

    hipLaunchKernelGGL(NLF_wt_kernel, dim3((H_ * H_ + 255) / 256), dim3(256), 0, stream,
                       W, wt);
    hipLaunchKernelGGL(NLF_ln_kernel, dim3(NPTS / 8), dim3(256), 0, stream,
                       inputs, gamma, beta, lnf);
    hipLaunchKernelGGL(NLF_gemm_kernel,
                       dim3((NPTS + WAVES_PER_WG * PTS_PER_WAVE - 1) /
                            (WAVES_PER_WG * PTS_PER_WAVE)),
                       dim3(256), 0, stream,
                       lnf, wt, W, bias, relative, nidx, inputs, out);
}
